// ATTN_86397562126707
// MI455X (gfx1250) — compile-verified, matrix operand order checked by reading, not run on hardware
//
#include <hip/hip_runtime.h>

typedef __attribute__((ext_vector_type(16))) _Float16 v16h;
typedef __attribute__((ext_vector_type(8)))  _Float16 v8h;
typedef __attribute__((ext_vector_type(8)))  float    v8f;

constexpr int Hh    = 16;
constexpr int DH    = 32;
constexpr int S_LEN = 2048;
constexpr int SPAD  = S_LEN + 32;   // zero-padded key dim (K rows, V^T cols)
constexpr int EPAD  = S_LEN + 48;   // zero-padded E rows (band reaches S+46)
constexpr int XSTR  = 3 * Hh * DH;  // 1536

__device__ __forceinline__ v8f wmma_f16(v16h a, v16h b, v8f c) {
  // D = A(16x32) x B(32x16) + C, f32 accumulate
  return __builtin_amdgcn_wmma_f32_16x16x32_f16(false, a, false, b, (short)0, c,
                                                false, false);
}

// A/B operand gather: `p` already includes the per-lane row and chunk (c0)
// displacement, so both halves are immediate-offset b128 loads.
// lanes 0-15 take dims {0..7, 16..23}, lanes 16-31 take dims {8..15, 24..31}.
__device__ __forceinline__ v16h load_tile2(const _Float16* __restrict__ p) {
  v8h lo = *(const v8h*)(p);
  v8h hi = *(const v8h*)(p + 16);
  return __builtin_shufflevector(lo, hi, 0, 1, 2, 3, 4, 5, 6, 7,
                                 8, 9, 10, 11, 12, 13, 14, 15);
}

// ---- Pre-pass A: present = stack(k,v); Kf16 row-major; V^T f16; all padded --
__global__ void attn_prep_x(const float* __restrict__ x,
                            float* __restrict__ present,
                            _Float16* __restrict__ kf,
                            _Float16* __restrict__ vt, int total) {
  int idx = blockIdx.x * blockDim.x + threadIdx.x;
  if (idx >= total) return;
  int d  = idx & (DH - 1);
  int s  = (idx / DH) % SPAD;
  int hb = idx / (DH * SPAD);
  int h  = hb % Hh;
  int b  = hb / Hh;

  float kval = 0.f, vval = 0.f;
  if (s < S_LEN) {
    const float* xr = x + ((size_t)b * S_LEN + s) * XSTR;
    kval = xr[Hh * DH + h * DH + d];
    vval = xr[2 * Hh * DH + h * DH + d];
    size_t pk = ((((size_t)b * 2 + 0) * Hh + h) * S_LEN + s) * DH + d;
    present[pk]                           = kval;
    present[pk + (size_t)Hh * S_LEN * DH] = vval;
  }
  kf[((size_t)(b * Hh + h) * SPAD + s) * DH + d] = (_Float16)kval;
  vt[((size_t)(b * Hh + h) * DH + d) * SPAD + s] = (_Float16)vval;
}

// ---- Pre-pass B: E -> f16, rows >= S zero-filled ----------------------------
__global__ void attn_prep_e(const float* __restrict__ E,
                            _Float16* __restrict__ ef, int total) {
  int idx = blockIdx.x * blockDim.x + threadIdx.x;
  if (idx >= total) return;
  int d = idx & (DH - 1);
  int m = (idx / DH) % EPAD;
  int h = idx / (DH * EPAD);
  float ev = (m < S_LEN) ? E[((size_t)h * S_LEN + m) * DH + d] : 0.f;
  ef[((size_t)h * EPAD + m) * DH + d] = (_Float16)ev;
}

// ---- Flash attention with skewed relative positions -------------------------
__global__ __launch_bounds__(32) void attn_fwd_kernel(
    const float* __restrict__ x, const _Float16* __restrict__ kf,
    const _Float16* __restrict__ ef, const _Float16* __restrict__ vt,
    float* __restrict__ out) {
  __shared__ float grel[48 * 16];  // G^T band tile (E-row x query), pitch 16

  const int lane = threadIdx.x & 31;
  const int col  = lane & 15;   // query column this lane owns in C/D layout
  const int hi   = lane >> 4;
  const int c0   = hi * 8;      // 16-bit operand chunk base
  const int q0   = blockIdx.x * 16;
  const int h    = blockIdx.y;
  const int b    = blockIdx.z;
  const int q    = q0 + col;

  // Fixed B operand: Q^T (only f32->f16 conversion left in this kernel)
  const float* qrow = x + ((size_t)b * S_LEN + q) * XSTR + h * DH;
  v16h bq;
#pragma unroll
  for (int i = 0; i < 8; i++) {
    bq[i]     = (_Float16)qrow[c0 + i];
    bq[8 + i] = (_Float16)qrow[c0 + 16 + i];
  }

  // Loop-carried base pointers (per-lane displacement folded in once).
  // All intra-iteration displacements are immediate offsets.
  const _Float16* kp = kf + (size_t)(b * Hh + h) * SPAD * DH
                          + (size_t)col * DH + c0;                    // += 32*DH
  const _Float16* ep = ef + (size_t)h * EPAD * DH
                          + (size_t)(S_LEN - 16 - q0 + col) * DH + c0; // += 32*DH
  const _Float16* vp = vt + ((size_t)(b * Hh + h) * DH + col) * SPAD + c0; // += 32

  v8f o0 = {};  // O^T rows d=0..15  (lane = query col)
  v8f o1 = {};  // O^T rows d=16..31
  float m_run = -3.0e38f, l_run = 0.f;
  // rsqrt(32) * log2(e): softmax in exp2 domain (v_exp_f32 is base-2)
  const float scale2 = 0.17677669529663687f * 1.4426950408889634f;

  const int jend = q0 + 16;  // causal: keys j <= q0+15
  for (int j0 = 0; j0 < jend; j0 += 32) {
    // prefetch next kv tile lines (speculative: silently dropped if invalid)
    __builtin_prefetch(kp + 32 * DH, 0, 1);
    __builtin_prefetch(ep + 32 * DH, 0, 1);
    __builtin_prefetch(vp + 32, 0, 1);

    // ---- content scores: S^T(16k x 16q) = K_tile x Q^T ---------------------
    v16h ak0 = load_tile2(kp);
    v16h ak1 = load_tile2(kp + 16 * DH);
    v8f zero = {};
    v8f s0 = wmma_f16(ak0, bq, zero);
    v8f s1 = wmma_f16(ak1, bq, zero);

    // ---- relative scores: G^T(48m x 16q) = E_band x Q^T (3 WMMAs) ----------
    // E row for (q,j): m = S-1-q+j = base + (15-qq) + jj ; pad rows read 0
    v8f g[3];
#pragma unroll
    for (int t = 0; t < 3; t++) {
      v16h ae = load_tile2(ep + t * 16 * DH);
      g[t] = wmma_f16(ae, bq, zero);
    }
    // D layout -> LDS: lane holds rows 16t+8*hi+r of its query column
#pragma unroll
    for (int r = 0; r < 8; r++) {
      grel[(8 * hi + r) * 16 + col]      = g[0][r];
      grel[(16 + 8 * hi + r) * 16 + col] = g[1][r];
      grel[(32 + 8 * hi + r) * 16 + col] = g[2][r];
    }
    __syncthreads();  // single-wave WG: free; orders ds writes before reads

    // ---- gather skewed rel, scale (log2 domain), causal mask ----------------
    float sl[8], sh[8];
#pragma unroll
    for (int r = 0; r < 8; r++) {
      int jj   = 8 * hi + r;
      int mloc = 15 - col + jj;  // 0..46
      float v0 = s0[r] + grel[mloc * 16 + col];
      float v1 = s1[r] + grel[(mloc + 16) * 16 + col];
      sl[r] = (j0 + jj      <= q) ? v0 * scale2 : -1.0e10f;
      sh[r] = (j0 + 16 + jj <= q) ? v1 * scale2 : -1.0e10f;
    }

    // ---- online softmax per query column (lane n <-> n+16 pair) -------------
    float mx = -3.0e38f;
#pragma unroll
    for (int r = 0; r < 8; r++) mx = fmaxf(mx, fmaxf(sl[r], sh[r]));
    mx = fmaxf(mx, __shfl_xor(mx, 16, 32));
    float m_new = fmaxf(m_run, mx);

    float pl[8], ph[8], lsum = 0.f;
#pragma unroll
    for (int r = 0; r < 8; r++) {
      pl[r] = exp2f(sl[r] - m_new);
      ph[r] = exp2f(sh[r] - m_new);
      lsum += pl[r] + ph[r];
    }
    lsum += __shfl_xor(lsum, 16, 32);
    float alpha = exp2f(m_run - m_new);
    l_run = l_run * alpha + lsum;
    m_run = m_new;
#pragma unroll
    for (int i = 0; i < 8; i++) { o0[i] *= alpha; o1[i] *= alpha; }

    // ---- P^T -> f16 B operand (pure in-lane) --------------------------------
    v16h bp;
#pragma unroll
    for (int i = 0; i < 8; i++) {
      bp[i]     = (_Float16)pl[i];
      bp[8 + i] = (_Float16)ph[i];
    }

    // ---- O^T += V^T_tile x P^T ---------------------------------------------
    v16h av0 = load_tile2(vp);
    v16h av1 = load_tile2(vp + 16 * SPAD);
    o0 = wmma_f16(av0, bp, o0);
    o1 = wmma_f16(av1, bp, o1);

    kp += 32 * DH;
    ep += 32 * DH;
    vp += 32;
    // no trailing barrier: same-wave LDS ops are in-order
  }

  // ---- write a[b, q, h*32 + d] = O^T / l -----------------------------------
  float inv_l = 1.f / l_run;
  float* orow = out + ((size_t)b * S_LEN + q) * (Hh * DH) + h * DH;
#pragma unroll
  for (int r = 0; r < 8; r++) {
    orow[8 * hi + r]      = o0[r] * inv_l;
    orow[16 + 8 * hi + r] = o1[r] * inv_l;
  }
}

extern "C" void kernel_launch(void* const* d_in, const int* in_sizes, int n_in,
                              void* d_out, int out_size, void* d_ws, size_t ws_size,
                              hipStream_t stream) {
  const float* x = (const float*)d_in[0];
  const float* E = (const float*)d_in[1];
  float* out = (float*)d_out;
  const int B = in_sizes[0] / (S_LEN * XSTR);

  float* present = out + (size_t)B * S_LEN * Hh * DH;  // after `a`

  // workspace carve-up (f16): V^T | K | E  (~10.7 MB total for B=2)
  _Float16* vt = (_Float16*)d_ws;                    // [B,H,DH,SPAD]
  _Float16* kf = vt + (size_t)B * Hh * DH * SPAD;    // [B,H,SPAD,DH]
  _Float16* ef = kf + (size_t)B * Hh * SPAD * DH;    // [H,EPAD,DH]

  int totx = B * Hh * SPAD * DH;
  attn_prep_x<<<(totx + 255) / 256, 256, 0, stream>>>(x, present, kf, vt, totx);
  int tote = Hh * EPAD * DH;
  attn_prep_e<<<(tote + 255) / 256, 256, 0, stream>>>(E, ef, tote);

  dim3 grid(S_LEN / 16, Hh, B);
  attn_fwd_kernel<<<grid, 32, 0, stream>>>(x, kf, ef, vt, out);
}
